// TPAAttention_13288628814265
// MI455X (gfx1250) — compile-verified, matrix operand order checked by reading, not run on hardware
//
#include <hip/hip_runtime.h>
#include <cstdint>
#include <cstddef>

#define BB      2
#define SS      2048
#define HH      2048
#define NH      16
#define NKV     8
#define HD      128
#define QR      6
#define KR      2
#define VR      2
#define PCOLS   1408
#define WINDOW  1024
#define SOFTCAPF 50.0f
#define SCALINGF 0.08838834764831845f   // 128^-0.5

typedef __attribute__((ext_vector_type(16))) __bf16 v16bf;
typedef __attribute__((ext_vector_type(8)))  float  v8f;
typedef int v4i __attribute__((ext_vector_type(4)));
typedef __attribute__((address_space(1))) v4i as1_v4i;   // global AS pointer target
typedef __attribute__((address_space(3))) v4i as3_v4i;   // LDS AS pointer target

union Frag16 { uint4 u[2]; v16bf v; };

#if __has_builtin(__builtin_amdgcn_global_load_async_to_lds_b128) && \
    __has_builtin(__builtin_amdgcn_s_wait_asynccnt)
#define HAVE_ASYNC_LDS 1
#else
#define HAVE_ASYNC_LDS 0
#endif

// Stage one 16B chunk global -> LDS.  CDNA5 path: GLOBAL_LOAD_ASYNC_TO_LDS_B128
// (ASYNCcnt-tracked, no VGPR round trip).  AS-qualified pointers built via
// integer casts: AS1 = 64-bit VA; AS3 = low 32 bits of the flat LDS address
// (ISA 10.2).
__device__ __forceinline__ void stage_b128(const unsigned short* g, unsigned short* l) {
#if HAVE_ASYNC_LDS
  as1_v4i* gp = (as1_v4i*)(unsigned long long)(uintptr_t)g;
  as3_v4i* lp = (as3_v4i*)(unsigned int)(unsigned long long)(uintptr_t)l;
  __builtin_amdgcn_global_load_async_to_lds_b128(gp, lp, 0, 0);
#else
  *reinterpret_cast<uint4*>(l) = *reinterpret_cast<const uint4*>(g);
#endif
}

__device__ __forceinline__ unsigned short f2bf(float f) {
  unsigned int u = __float_as_uint(f);
  u += 0x7FFFu + ((u >> 16) & 1u);          // round-to-nearest-even
  return (unsigned short)(u >> 16);
}

__device__ __forceinline__ v8f wmma_bf16(const Frag16& a, const Frag16& b, v8f c) {
  return __builtin_amdgcn_wmma_f32_16x16x32_bf16(
      false, a.v, false, b.v, (short)0, c, false, false);
}

// ---------------------------------------------------------------------------
// Cast kernels: fp32 -> bf16 (plain and transposed-with-ld)
// ---------------------------------------------------------------------------
__global__ void cast_bf16_kernel(const float* __restrict__ src,
                                 unsigned short* __restrict__ dst, int n) {
  int i = blockIdx.x * blockDim.x + threadIdx.x;
  if (i < n) dst[i] = f2bf(src[i]);
}

// src[R][C] row-major -> dst[c*ldT + r]   (i.e. dst = src^T with row stride ldT)
__global__ void cast_bf16_T_kernel(const float* __restrict__ src,
                                   unsigned short* __restrict__ dst,
                                   int R, int C, int ldT) {
  int i = blockIdx.x * blockDim.x + threadIdx.x;
  if (i < R * C) {
    int r = i / C, c = i % C;
    dst[(size_t)c * ldT + r] = f2bf(src[(size_t)r * C + c]);
  }
}

// ---------------------------------------------------------------------------
// bf16 WMMA GEMM, K fixed at 2048: C[M,N] fp32 = A[M,2048] @ B with B given
// transposed as Bt[N,2048], both row-major bf16 (ld = 2048).
// 128x128 block tile, Kstep=32, double-buffered LDS staged with
// GLOBAL_LOAD_ASYNC_TO_LDS_B128: tile t+1 streams in while tile t computes;
// s_wait_asynccnt(4) drains only tile t (in-order completion, 4 chunks/thread).
// 8 waves (4 M-strips x 2 N-strips), 8 wmma per wave per K-step.
// ---------------------------------------------------------------------------
#define KDIM  2048
#define KSTEP 32
#define LDSL  40     // 32 + 8 pad; 80B row stride keeps 16B chunk alignment
#define NKT   (KDIM / KSTEP)

__global__ __launch_bounds__(256) void gemm_bf16_wmma(
    const unsigned short* __restrict__ A, const unsigned short* __restrict__ Bt,
    float* __restrict__ C, int M, int N, int ldC) {
  __shared__ __align__(16) unsigned short sA[2][128][LDSL];
  __shared__ __align__(16) unsigned short sB[2][128][LDSL];
  const int tid  = threadIdx.x;
  const int lane = tid & 31;
  const int wid  = tid >> 5;
  const int wm   = wid & 3;        // 4 strips of 32 rows
  const int wn   = wid >> 2;       // 2 strips of 64 cols
  const int m0 = blockIdx.y * 128;
  const int n0 = blockIdx.x * 128;

  // this thread's 2 chunks per matrix per tile: 512 chunks = 128 rows x 4
  const int c0r  = (tid * 2) >> 2;            // row of chunk 0 (chunk 1 same row or next)
  const int c0s  = ((tid * 2) & 3) << 3;
  const int c1r  = (tid * 2 + 1) >> 2;
  const int c1s  = ((tid * 2 + 1) & 3) << 3;
  const unsigned short* gA0 = A  + (size_t)(m0 + c0r) * KDIM + c0s;
  const unsigned short* gA1 = A  + (size_t)(m0 + c1r) * KDIM + c1s;
  const unsigned short* gB0 = Bt + (size_t)(n0 + c0r) * KDIM + c0s;
  const unsigned short* gB1 = Bt + (size_t)(n0 + c1r) * KDIM + c1s;

  const v8f vzero = {0.f,0.f,0.f,0.f,0.f,0.f,0.f,0.f};
  v8f acc[2][4];
#pragma unroll
  for (int mi = 0; mi < 2; ++mi)
#pragma unroll
    for (int ni = 0; ni < 4; ++ni) acc[mi][ni] = vzero;

  const int r16 = lane & 15;
  const int kb8 = (lane >> 4) << 3;   // CDNA5 16-bit A/B fragment K-swizzle

  // prologue: stage tile 0 into buffer 0
  stage_b128(gA0, &sA[0][c0r][c0s]);
  stage_b128(gA1, &sA[0][c1r][c1s]);
  stage_b128(gB0, &sB[0][c0r][c0s]);
  stage_b128(gB1, &sB[0][c1r][c1s]);

#pragma unroll 2
  for (int t = 0; t < NKT; ++t) {
    const int buf = t & 1;
    if (t + 1 < NKT) {
      __syncthreads();               // readers of buf^1 (iter t-1) are done
      const int k1 = (t + 1) * KSTEP;
      stage_b128(gA0 + k1, &sA[buf ^ 1][c0r][c0s]);
      stage_b128(gA1 + k1, &sA[buf ^ 1][c1r][c1s]);
      stage_b128(gB0 + k1, &sB[buf ^ 1][c0r][c0s]);
      stage_b128(gB1 + k1, &sB[buf ^ 1][c1r][c1s]);
#if HAVE_ASYNC_LDS
      __builtin_amdgcn_s_wait_asynccnt(4);   // tile t resident; t+1 in flight
#endif
    } else {
#if HAVE_ASYNC_LDS
      __builtin_amdgcn_s_wait_asynccnt(0);
#endif
    }
    __syncthreads();                 // tile t visible to all waves

    Frag16 af[2], bf[4];
#pragma unroll
    for (int mi = 0; mi < 2; ++mi) {
      const unsigned short* base = &sA[buf][wm * 32 + mi * 16][0];
      af[mi].u[0] = *reinterpret_cast<const uint4*>(base + r16 * LDSL + kb8);
      af[mi].u[1] = *reinterpret_cast<const uint4*>(base + r16 * LDSL + kb8 + 16);
    }
#pragma unroll
    for (int ni = 0; ni < 4; ++ni) {
      const unsigned short* base = &sB[buf][wn * 64 + ni * 16][0];
      bf[ni].u[0] = *reinterpret_cast<const uint4*>(base + r16 * LDSL + kb8);
      bf[ni].u[1] = *reinterpret_cast<const uint4*>(base + r16 * LDSL + kb8 + 16);
    }
#pragma unroll
    for (int mi = 0; mi < 2; ++mi)
#pragma unroll
      for (int ni = 0; ni < 4; ++ni)
        acc[mi][ni] = wmma_bf16(af[mi], bf[ni], acc[mi][ni]);
  }

  const int g = lane >> 4;
#pragma unroll
  for (int mi = 0; mi < 2; ++mi)
#pragma unroll
    for (int ni = 0; ni < 4; ++ni)
#pragma unroll
      for (int v = 0; v < 8; ++v) {
        int r  = m0 + wm * 32 + mi * 16 + g * 8 + v;   // C/D frag row mapping
        int cc = n0 + wn * 64 + ni * 16 + r16;
        C[(size_t)r * ldC + cc] = acc[mi][ni][v];
      }
}

// ---------------------------------------------------------------------------
// RoPE + rank contraction: P[b,s,1408] fp32 -> q[B,NH,S,HD], k[B,NKV,S,HD],
// vT[B,NKV,HD,S] all bf16.  One block per (b,s); thread = head-dim d.
// Column layout of P: Aq[0,96) Ak[96,112) Av[112,128) Bq[128,896) Bk[896,1152) Bv[1152,1408)
// ---------------------------------------------------------------------------
__global__ __launch_bounds__(128) void rope_contract(
    const float* __restrict__ P, const float* __restrict__ cosb,
    const float* __restrict__ sinb, unsigned short* __restrict__ qo,
    unsigned short* __restrict__ ko, unsigned short* __restrict__ vto) {
  __shared__ float sAhead[128];
  const int bs = blockIdx.x;
  const int b = bs / SS;
  const int s = bs % SS;
  const float* pr = P + (size_t)bs * PCOLS;
  const int d = threadIdx.x;
  sAhead[d] = pr[d];                 // A_q | A_k | A_v = first 128 columns
  __syncthreads();

  const int dl = d & 63;
  const bool hi = d >= 64;
  const float c  = cosb[(size_t)s * 64 + dl];
  const float sn = sinb[(size_t)s * 64 + dl];

  float rq[QR];
#pragma unroll
  for (int r = 0; r < QR; ++r) {
    float a  = pr[128 + r * HD + dl];
    float b2 = pr[128 + r * HD + dl + 64];
    rq[r] = hi ? (a * sn + b2 * c) : (a * c - b2 * sn);
  }
  float rk[KR];
#pragma unroll
  for (int r = 0; r < KR; ++r) {
    float a  = pr[896 + r * HD + dl];
    float b2 = pr[896 + r * HD + dl + 64];
    rk[r] = hi ? (a * sn + b2 * c) : (a * c - b2 * sn);
  }
  float bv[VR];
#pragma unroll
  for (int r = 0; r < VR; ++r) bv[r] = pr[1152 + r * HD + d];

#pragma unroll
  for (int h = 0; h < NH; ++h) {
    float acc = 0.f;
#pragma unroll
    for (int r = 0; r < QR; ++r) acc += sAhead[h * QR + r] * rq[r];
    qo[(((size_t)b * NH + h) * SS + s) * HD + d] = f2bf(acc * (1.0f / QR));
  }
#pragma unroll
  for (int h = 0; h < NKV; ++h) {
    float acck = 0.f, accv = 0.f;
#pragma unroll
    for (int r = 0; r < KR; ++r) acck += sAhead[96 + h * KR + r] * rk[r];
#pragma unroll
    for (int r = 0; r < VR; ++r) accv += sAhead[112 + h * VR + r] * bv[r];
    ko[(((size_t)b * NKV + h) * SS + s) * HD + d]  = f2bf(acck * 0.5f);
    vto[(((size_t)b * NKV + h) * HD + d) * SS + s] = f2bf(accv * 0.5f);  // V^T
  }
}

// ---------------------------------------------------------------------------
// Flash attention with WMMA.  One wave (32 thr) per (b, h, 16 q-rows).
// k-tiles of 32: 8 wmma QK^T, softcap+window mask, online softmax (16-lane
// shuffles), probs->bf16 via 1KB LDS re-swizzle, 8 wmma PV into 16x128 acc.
// ---------------------------------------------------------------------------
__global__ __launch_bounds__(32) void tpa_attn_wmma(
    const unsigned short* __restrict__ qbuf, const unsigned short* __restrict__ kbuf,
    const unsigned short* __restrict__ vtbuf, unsigned short* __restrict__ attn) {
  __shared__ __align__(16) unsigned short sP[16][40];
  const int lane = threadIdx.x & 31;
  const int bid = blockIdx.x;
  const int qt = bid & 127;          // S/16 = 128
  const int h  = (bid >> 7) & 15;
  const int b  = bid >> 11;
  const int q0 = qt * 16;

  const unsigned short* qb = qbuf + (((size_t)b * NH + h) * SS + q0) * HD;
  const unsigned short* kp = kbuf + (((size_t)b * NKV + (h >> 1)) * SS) * HD;
  const unsigned short* vt = vtbuf + (((size_t)b * NKV + (h >> 1)) * HD) * SS;

  const int r16 = lane & 15;
  const int g   = lane >> 4;
  const int kb8 = g << 3;

  Frag16 qf[4];                      // 16 rows x 128 d, resident in VGPRs
#pragma unroll
  for (int j = 0; j < 4; ++j) {
    const unsigned short* p = qb + (size_t)r16 * HD + j * 32 + kb8;
    qf[j].u[0] = *reinterpret_cast<const uint4*>(p);
    qf[j].u[1] = *reinterpret_cast<const uint4*>(p + 16);
  }

  const v8f vzero = {0.f,0.f,0.f,0.f,0.f,0.f,0.f,0.f};
  v8f of[8];
#pragma unroll
  for (int f = 0; f < 8; ++f) of[f] = vzero;
  float mstate[8], lstate[8];
#pragma unroll
  for (int v = 0; v < 8; ++v) { mstate[v] = -3.0e38f; lstate[v] = 0.f; }

  int kstart = q0 - (WINDOW - 1);
  if (kstart < 0) kstart = 0;
  kstart &= ~31;
  const int kend = q0 + 15;
  const float capmul = SCALINGF / SOFTCAPF;

  for (int s0 = kstart; s0 <= kend; s0 += 32) {
    v8f c0 = vzero, c1 = vzero;
#pragma unroll
    for (int j = 0; j < 4; ++j) {    // K-dim = HD = 128 -> 4 wmma per n-tile
      Frag16 kf;
      const unsigned short* p0 = kp + (size_t)(s0 + r16) * HD + j * 32 + kb8;
      kf.u[0] = *reinterpret_cast<const uint4*>(p0);
      kf.u[1] = *reinterpret_cast<const uint4*>(p0 + 16);
      c0 = wmma_bf16(qf[j], kf, c0);
      const unsigned short* p1 = kp + (size_t)(s0 + 16 + r16) * HD + j * 32 + kb8;
      kf.u[0] = *reinterpret_cast<const uint4*>(p1);
      kf.u[1] = *reinterpret_cast<const uint4*>(p1 + 16);
      c1 = wmma_bf16(qf[j], kf, c1);
    }

    float p0a[8], p1a[8], alpha[8];
#pragma unroll
    for (int v = 0; v < 8; ++v) {
      int mrow = q0 + g * 8 + v;
      int col0 = s0 + r16;
      int col1 = col0 + 16;
      float s0v = SOFTCAPF * tanhf(c0[v] * capmul);
      float s1v = SOFTCAPF * tanhf(c1[v] * capmul);
      if (!((mrow >= col0) && (mrow - col0 < WINDOW))) s0v = -3.0e38f;
      if (!((mrow >= col1) && (mrow - col1 < WINDOW))) s1v = -3.0e38f;
      float rm = fmaxf(s0v, s1v);
#pragma unroll
      for (int off = 1; off < 16; off <<= 1)
        rm = fmaxf(rm, __shfl_xor(rm, off, 32));   // stays inside 16-lane group
      float mold = mstate[v];
      float mnew = fmaxf(mold, rm);
      float msafe = fmaxf(mnew, -1.0e30f);         // fully-masked-tile guard
      float e0 = __expf(s0v - msafe);
      float e1 = __expf(s1v - msafe);
      float rs = e0 + e1;
#pragma unroll
      for (int off = 1; off < 16; off <<= 1) rs += __shfl_xor(rs, off, 32);
      float a = __expf(mold - mnew);
      lstate[v] = lstate[v] * a + rs;
      mstate[v] = mnew;
      alpha[v] = a;
      p0a[v] = e0; p1a[v] = e1;
    }
#pragma unroll
    for (int f = 0; f < 8; ++f)
#pragma unroll
      for (int v = 0; v < 8; ++v) of[f][v] *= alpha[v];

    // probs: D-frag layout -> A-frag layout via LDS (bf16)
#pragma unroll
    for (int v = 0; v < 8; ++v) {
      sP[g * 8 + v][r16]      = f2bf(p0a[v]);
      sP[g * 8 + v][16 + r16] = f2bf(p1a[v]);
    }
    __syncthreads();
    Frag16 pf;
    {
      const unsigned short* base = &sP[0][0];
      pf.u[0] = *reinterpret_cast<const uint4*>(base + r16 * 40 + kb8);
      pf.u[1] = *reinterpret_cast<const uint4*>(base + r16 * 40 + kb8 + 16);
    }
    __syncthreads();

#pragma unroll
    for (int f = 0; f < 8; ++f) {    // N = d tiles; B-operand direct from V^T
      Frag16 vf;
      const unsigned short* p = vt + (size_t)(f * 16 + r16) * SS + s0 + kb8;
      vf.u[0] = *reinterpret_cast<const uint4*>(p);
      vf.u[1] = *reinterpret_cast<const uint4*>(p + 16);
      of[f] = wmma_bf16(pf, vf, of[f]);
    }
  }

#pragma unroll
  for (int f = 0; f < 8; ++f)
#pragma unroll
    for (int v = 0; v < 8; ++v) {
      float val = of[f][v] / lstate[v];
      size_t srow = (size_t)b * SS + q0 + g * 8 + v;
      attn[srow * (NH * HD) + h * HD + f * 16 + r16] = f2bf(val);
    }
}

// ---------------------------------------------------------------------------
extern "C" void kernel_launch(void* const* d_in, const int* in_sizes, int n_in,
                              void* d_out, int out_size, void* d_ws, size_t ws_size,
                              hipStream_t stream) {
  const float* hidden = (const float*)d_in[0];
  const float* fcos   = (const float*)d_in[1];
  const float* fsin   = (const float*)d_in[2];
  // d_in[3] kv_write_indices (identity), d_in[4] mask (recomputed analytically)
  const float* W_A_q = (const float*)d_in[5];
  const float* W_A_k = (const float*)d_in[6];
  const float* W_A_v = (const float*)d_in[7];
  const float* W_B_q = (const float*)d_in[8];
  const float* W_B_k = (const float*)d_in[9];
  const float* W_B_v = (const float*)d_in[10];
  const float* W_o   = (const float*)d_in[11];
  float* out = (float*)d_out;

  char* wsb = (char*)d_ws;
  size_t off = 0;
  auto take = [&](size_t bytes) {
    char* p = wsb + off;
    off = (off + bytes + 255) & ~(size_t)255;
    return p;
  };
  unsigned short* Xbf    = (unsigned short*)take((size_t)BB * SS * HH * 2);
  unsigned short* WcatT  = (unsigned short*)take((size_t)PCOLS * HH * 2);
  unsigned short* WoT    = (unsigned short*)take((size_t)HH * HH * 2);
  float*          Pbuf   = (float*)take((size_t)BB * SS * PCOLS * 4);
  unsigned short* qbf    = (unsigned short*)take((size_t)BB * NH * SS * HD * 2);
  unsigned short* kbf    = (unsigned short*)take((size_t)BB * NKV * SS * HD * 2);
  unsigned short* vtbf   = (unsigned short*)take((size_t)BB * NKV * HD * SS * 2);
  unsigned short* attnbf = (unsigned short*)take((size_t)BB * SS * NH * HD * 2);

  { int n = BB * SS * HH;
    cast_bf16_kernel<<<(n + 255) / 256, 256, 0, stream>>>(hidden, Xbf, n); }

  struct WInfo { const float* w; int cols; int coloff; };
  const WInfo winfo[6] = {
    {W_A_q,  96,    0}, {W_A_k,  16,   96}, {W_A_v,  16,  112},
    {W_B_q, 768,  128}, {W_B_k, 256,  896}, {W_B_v, 256, 1152} };
  for (int i = 0; i < 6; ++i) {
    int n = HH * winfo[i].cols;
    cast_bf16_T_kernel<<<(n + 255) / 256, 256, 0, stream>>>(
        winfo[i].w, WcatT + (size_t)winfo[i].coloff * HH, HH, winfo[i].cols, HH);
  }
  { int n = HH * HH;
    cast_bf16_T_kernel<<<(n + 255) / 256, 256, 0, stream>>>(W_o, WoT, HH, HH, HH); }

  // P = X @ Wcat  (fused A_q|A_k|A_v|B_q|B_k|B_v projection)
  gemm_bf16_wmma<<<dim3(PCOLS / 128, (BB * SS) / 128), 256, 0, stream>>>(
      Xbf, WcatT, Pbuf, BB * SS, PCOLS, PCOLS);

  rope_contract<<<BB * SS, 128, 0, stream>>>(Pbuf, fcos, fsin, qbf, kbf, vtbf);

  tpa_attn_wmma<<<BB * NH * (SS / 16), 32, 0, stream>>>(qbf, kbf, vtbf, attnbf);

  // out = attn @ W_o
  gemm_bf16_wmma<<<dim3((NH * HD) / 128, (BB * SS) / 128), 256, 0, stream>>>(
      attnbf, WoT, out, BB * SS, NH * HD, NH * HD);

  (void)in_sizes; (void)n_in; (void)out_size; (void)ws_size;
}
